// Decoder_25228637897220
// MI455X (gfx1250) — matrix operand order repaired and checked by reading, not compiled, not run on hardware
//
#include <hip/hip_runtime.h>
#include <hip/hip_bf16.h>

// ---------------------------------------------------------------------------
// Fused Social-GAN LSTM decoder for MI455X (gfx1250), v2.
//
//   gates = h @ w_hh^T + r @ (w_ih @ w_emb)^T + (b_ih + b_hh + w_ih @ b_emb)
//
// v2 changes vs v1 (LDS-bandwidth + VALU driven):
//  * Wave-per-gate-quad split: each of the 8 waves owns 16 gate channels
//    (i/f/g/o) for all 128 rows -> its 16 w_hh B-fragments are cached in
//    128 VGPRs ONCE for all 32 timesteps (zero B-matrix LDS traffic in loop).
//  * h state double-buffered in LDS (waves now write columns / read rows).
//  * LSTM nonlinearity uses gfx1250 v_tanh_f32 (inline asm, TRANS hazard
//    covered by s_delay_alu TRANS32_DEP_1); sigmoid = 0.5+0.5*tanh(x/2).
//  * rel = h2 @ w_pos^T uses v_dot2_f32_f16 (__builtin_amdgcn_fdot2).
// ---------------------------------------------------------------------------

typedef __attribute__((ext_vector_type(16))) _Float16 v16h;
typedef __attribute__((ext_vector_type(2)))  _Float16 v2h;
typedef __attribute__((ext_vector_type(8)))  float    v8f;

#define BATCH        65536
#define H            128
#define EMB          64
#define SEQ          32
#define ROWS_PER_WG  128
#define THREADS      256
#define AP           136   // padded LDS pitch in halves (68 dwords -> bank stride 4)

// LDS layout (bytes)
#define OFF_W    0                          // w_hh f16: 512*AP*2 = 139264
#define OFF_A0   (512 * AP * 2)             // h buf 0:  128*AP*2 =  34816
#define OFF_A1   (OFF_A0 + 128 * AP * 2)    // h buf 1:              34816
#define OFF_WC   (OFF_A1 + 128 * AP * 2)    // wc f32: 512*2*4 = 4096
#define OFF_B    (OFF_WC + 4096)            // bias f32: 512*4 = 2048
#define OFF_WPH  (OFF_B + 2048)             // w_pos f16: 256*2 = 512
#define OFF_BP   (OFF_WPH + 512)            // b_pos f32 (pad 16)
#define OFF_REL  (OFF_BP + 16)              // rel scratch f32: 256*4 = 1024
#define LDS_BYTES (OFF_REL + 1024)          // 216592 bytes (< 320 KB / WGP)

__device__ __forceinline__ float tanh_hw(float x) {
  float r;
  asm("v_tanh_f32 %0, %1\n\ts_delay_alu instid0(TRANS32_DEP_1)"
      : "=v"(r) : "v"(x));
  return r;
}
__device__ __forceinline__ float sig_hw(float x) {
  return fmaf(0.5f, tanh_hw(0.5f * x), 0.5f);
}

// B operand (w_hh^T column n): 16 contiguous halves per lane.
// halves [0..15] = K = kt*32 + 16*(lane/16) + {0..15} of column n.
__device__ __forceinline__ v16h lds_fragB(const _Float16* sW, int n, int kt, int half) {
  union { v16h v; uint4 u[2]; } t;
  const char* p = (const char*)(sW + n * AP) + kt * 64 + half * 32;
  t.u[0] = *(const uint4*)(p);
  t.u[1] = *(const uint4*)(p + 16);
  return t.v;
}

__global__ __launch_bounds__(THREADS, 1)
void decoder_lstm_fused(const float* __restrict__ last_pos_rel,
                        const float* __restrict__ hh,
                        const float* __restrict__ ch,
                        const float* __restrict__ w_ih,
                        const float* __restrict__ w_hh,
                        const float* __restrict__ b_ih,
                        const float* __restrict__ b_hh,
                        const float* __restrict__ w_emb,
                        const float* __restrict__ b_emb,
                        const float* __restrict__ w_pos,
                        const float* __restrict__ b_pos,
                        float* __restrict__ out)
{
  extern __shared__ char smem[];
  _Float16* sW   = (_Float16*)(smem + OFF_W);
  _Float16* sA0  = (_Float16*)(smem + OFF_A0);
  _Float16* sA1  = (_Float16*)(smem + OFF_A1);
  float*    sWC  = (float*)(smem + OFF_WC);
  float*    sBias= (float*)(smem + OFF_B);
  _Float16* sWPh = (_Float16*)(smem + OFF_WPH);
  float*    sBP  = (float*)(smem + OFF_BP);
  float*    sREL = (float*)(smem + OFF_REL);

  const int tid  = threadIdx.x;
  const int wv   = tid >> 5;
  const int lane = tid & 31;
  const int ln   = lane & 15;
  const int half = lane >> 4;
  const int row0 = blockIdx.x * ROWS_PER_WG;

  // ---------------- cooperative init ----------------
  for (int idx = tid; idx < 4 * H * H; idx += THREADS)              // w_hh -> f16
    sW[(idx >> 7) * AP + (idx & 127)] = (_Float16)w_hh[idx];

  for (int n = tid; n < 4 * H; n += THREADS) {                      // folded embedding
    const float* wr = &w_ih[n * EMB];
    float c0 = 0.f, c1 = 0.f, bb = b_ih[n] + b_hh[n];
    for (int e = 0; e < EMB; ++e) {
      float w = wr[e];
      c0 += w * w_emb[e * 2 + 0];
      c1 += w * w_emb[e * 2 + 1];
      bb += w * b_emb[e];
    }
    sWC[n * 2 + 0] = c0; sWC[n * 2 + 1] = c1; sBias[n] = bb;
  }
  for (int idx = tid; idx < 2 * H; idx += THREADS) sWPh[idx] = (_Float16)w_pos[idx];
  if (tid < 2) sBP[tid] = b_pos[tid];

  for (int idx = tid; idx < ROWS_PER_WG * H; idx += THREADS) {      // h0 -> buffer 0
    int m = idx >> 7, k = idx & 127;
    sA0[m * AP + k] = (_Float16)hh[(size_t)(row0 + m) * H + k];
  }
  for (int idx = tid; idx < ROWS_PER_WG * 2; idx += THREADS)        // r0 = last_pos_rel
    sREL[idx] = last_pos_rel[(size_t)row0 * 2 + idx];
  __syncthreads();

  // ---------------- per-wave loop-invariant registers ----------------
  const int ch0 = wv * 16 + ln;                 // this lane's gate channel
  const float bI = sBias[ch0],        bF = sBias[ch0 + 128];
  const float bG = sBias[ch0 + 256],  bO = sBias[ch0 + 384];
  const float wI0 = sWC[ch0 * 2],           wI1 = sWC[ch0 * 2 + 1];
  const float wF0 = sWC[(ch0 + 128) * 2],   wF1 = sWC[(ch0 + 128) * 2 + 1];
  const float wG0 = sWC[(ch0 + 256) * 2],   wG1 = sWC[(ch0 + 256) * 2 + 1];
  const float wO0 = sWC[(ch0 + 384) * 2],   wO1 = sWC[(ch0 + 384) * 2 + 1];

  v16h bc[4][4];                                // B cache: 4 gates x 4 k-tiles (128 VGPRs)
  #pragma unroll
  for (int g = 0; g < 4; ++g)
    #pragma unroll
    for (int kt = 0; kt < 4; ++kt)
      bc[g][kt] = lds_fragB(sW, g * 128 + ch0, kt, half);

  // c state: element (mt,v) = c[row0 + mt*16 + v + 8*half][ch0]
  float c_reg[8][8];
  #pragma unroll
  for (int mt = 0; mt < 8; ++mt)
    #pragma unroll
    for (int v = 0; v < 8; ++v)
      c_reg[mt][v] = ch[(size_t)(row0 + mt * 16 + v + 8 * half) * H + ch0];

  // ---------------- 32 fused timesteps ----------------
  for (int s = 0; s < SEQ; ++s) {
    _Float16* sAr = (s & 1) ? sA1 : sA0;        // read h
    _Float16* sAw = (s & 1) ? sA0 : sA1;        // write h2

    #pragma unroll
    for (int mt = 0; mt < 8; ++mt) {
      // A fragments for rows mt*16 .. mt*16+15 (contiguous k = kt*32 + 16*half + {0..15}, same order as B)
      const char* aBase = (const char*)(sAr + (mt * 16 + ln) * AP);
      v16h af[4];
      #pragma unroll
      for (int kt = 0; kt < 4; ++kt) {
        union { v16h v; uint4 u[2]; } t;
        const char* p = aBase + kt * 64 + half * 32;
        t.u[0] = *(const uint4*)(p);
        t.u[1] = *(const uint4*)(p + 16);
        af[kt] = t.v;
      }

      v8f aI, aF, aG, aO;
      #pragma unroll
      for (int v = 0; v < 8; ++v) { aI[v] = bI; aF[v] = bF; aG[v] = bG; aO[v] = bO; }

      #pragma unroll
      for (int kt = 0; kt < 4; ++kt) {
        aI = __builtin_amdgcn_wmma_f32_16x16x32_f16(false, af[kt], false, bc[0][kt], (short)0, aI, false, false);
        aF = __builtin_amdgcn_wmma_f32_16x16x32_f16(false, af[kt], false, bc[1][kt], (short)0, aF, false, false);
        aG = __builtin_amdgcn_wmma_f32_16x16x32_f16(false, af[kt], false, bc[2][kt], (short)0, aG, false, false);
        aO = __builtin_amdgcn_wmma_f32_16x16x32_f16(false, af[kt], false, bc[3][kt], (short)0, aO, false, false);
      }

      // rank-2 folded-embedding epilogue + LSTM nonlinearity (v_tanh_f32)
      _Float16* hdst = sAw + (mt * 16 + 8 * half) * AP + ch0;
      #pragma unroll
      for (int v = 0; v < 8; ++v) {
        const float2 r = *(const float2*)(sREL + (mt * 16 + v + 8 * half) * 2);
        const float gi = aI[v] + r.x * wI0 + r.y * wI1;
        const float gf = aF[v] + r.x * wF0 + r.y * wF1;
        const float gg = aG[v] + r.x * wG0 + r.y * wG1;
        const float go = aO[v] + r.x * wO0 + r.y * wO1;
        const float cn = sig_hw(gf) * c_reg[mt][v] + sig_hw(gi) * tanh_hw(gg);
        c_reg[mt][v] = cn;
        hdst[v * AP] = (_Float16)(sig_hw(go) * tanh_hw(cn));
      }
    }
    __syncthreads();   // h2 (all columns) + previous sREL consumption complete

    // rel = h2 @ w_pos^T + b_pos : lane (ln,half) of wave wv -> row wv*16+ln, comp half
    {
      const _Float16* hp  = sAw + (wv * 16 + ln) * AP;
      const _Float16* wph = sWPh + half * H;
      float acc = sBP[half];
      #pragma unroll
      for (int k = 0; k < H; k += 2) {
        v2h a2 = *(const v2h*)(hp + k);
        v2h w2 = *(const v2h*)(wph + k);
#if __has_builtin(__builtin_amdgcn_fdot2)
        acc = __builtin_amdgcn_fdot2(a2, w2, acc, false);
#else
        acc += (float)a2[0] * (float)w2[0] + (float)a2[1] * (float)w2[1];
#endif
      }
      out[((size_t)s * BATCH + row0 + wv * 16 + ln) * 2 + half] = acc;
      sREL[(wv * 16 + ln) * 2 + half] = acc;
    }
    __syncthreads();   // sREL visible for next step's epilogue
  }

  // ---------------- hN output (final h lives in buffer 0 after 32 steps) ----
  const size_t HN = (size_t)SEQ * BATCH * 2;
  for (int idx = tid; idx < ROWS_PER_WG * H; idx += THREADS) {
    int m = idx >> 7, k = idx & 127;
    out[HN + (size_t)(row0 + m) * H + k] = (float)sA0[m * AP + k];
  }
}

extern "C" void kernel_launch(void* const* d_in, const int* in_sizes, int n_in,
                              void* d_out, int out_size, void* d_ws, size_t ws_size,
                              hipStream_t stream) {
  (void)in_sizes; (void)n_in; (void)out_size; (void)d_ws; (void)ws_size;
  const float* last_pos_rel = (const float*)d_in[1];
  const float* hh    = (const float*)d_in[2];
  const float* ch    = (const float*)d_in[3];
  // d_in[4] seq_start_end unused (pool_every_timestep=False); d_in[0] last_pos is dead state
  const float* w_ih  = (const float*)d_in[5];
  const float* w_hh  = (const float*)d_in[6];
  const float* b_ih  = (const float*)d_in[7];
  const float* b_hh  = (const float*)d_in[8];
  const float* w_emb = (const float*)d_in[9];
  const float* b_emb = (const float*)d_in[10];
  const float* w_pos = (const float*)d_in[11];
  const float* b_pos = (const float*)d_in[12];
  float* out = (float*)d_out;

  dim3 grid(BATCH / ROWS_PER_WG);   // 512 persistent workgroups
  dim3 block(THREADS);              // 8 wave32
  decoder_lstm_fused<<<grid, block, LDS_BYTES, stream>>>(
      last_pos_rel, hh, ch, w_ih, w_hh, b_ih, b_hh,
      w_emb, b_emb, w_pos, b_pos, out);
}
